// SequenceFlowBlock_38302518346489
// MI455X (gfx1250) — compile-verified, matrix operand order checked by reading, not run on hardware
//
#include <hip/hip_runtime.h>
#include <hip/hip_bf16.h>
#include <cmath>

// ---------------------------------------------------------------------------
// MMDiT dual-stream block for MI455X (gfx1250, wave32, WMMA).
// All GEMMs run on v_wmma_f32_16x16x32_f16 (f16 in, f32 accumulate).
// Attention is flash-style (online softmax); softmax row reductions use
// DPP16 row_ror (VALU) instead of ds_bpermute.
// ---------------------------------------------------------------------------

typedef _Float16 f16;
typedef __attribute__((ext_vector_type(16))) _Float16 v16h;
typedef __attribute__((ext_vector_type(8)))  _Float16 v8h;
typedef __attribute__((ext_vector_type(8)))  float    v8f;

namespace cfg {
constexpr int DIM   = 512;
constexpr int SEQ   = 2048;
constexpr int BATCH = 4;
constexpr int HEADS = 8;
constexpr int DHEAD = 64;
constexpr int ROWS  = BATCH * SEQ;          // 8192
constexpr int MODC  = 6 * DIM;              // 3072
}

// ---------------------------------------------------------------------------
// WMMA fragment helpers (ISA 7.12.2 layouts, wave32).
// 16-bit A (16x32 MxK): lane m=lane&15, hi=lane>>4. Fragment = two contiguous
// 8-f16 chunks at k = 8*hi and k = 16+8*hi.  B (32x16 KxN) mirrors with
// n = lane&15 when stored K-major per column ("Bt[n][k]").
// ---------------------------------------------------------------------------
__device__ __forceinline__ v16h load_frag(const f16* __restrict__ p, int hi) {
  union { v16h v; v8h h[2]; } u;
  u.h[0] = *(const v8h*)(p + 8 * hi);
  u.h[1] = *(const v8h*)(p + 16 + 8 * hi);
  return u.v;
}
// B fragment where only k=0..15 are valid (K padded with zeros in A).
__device__ __forceinline__ v16h load_frag_k16(const f16* __restrict__ p, int hi) {
  union { v16h v; v8h h[2]; } u;
  v8h z = {};
  u.h[0] = *(const v8h*)(p + 8 * hi);
  u.h[1] = z;
  return u.v;
}

__device__ __forceinline__ v8f wmma_f16(v16h a, v16h b, v8f c) {
  return __builtin_amdgcn_wmma_f32_16x16x32_f16(false, a, false, b, (short)0, c,
                                                false, false);
}

// ---------------------------------------------------------------------------
// DPP16 row-rotate cross-lane helpers: rotation within each 16-lane row.
// row_ror:n = 0x120 | n. Pure VALU (v_mov_b32_dpp), no LDS, no dscnt waits.
// ---------------------------------------------------------------------------
template <int N>
__device__ __forceinline__ float dpp_ror(float v) {
  union { float f; int i; } a, r;
  a.f = v;
  r.i = __builtin_amdgcn_update_dpp(0, a.i, 0x120 | N, 0xf, 0xf, true);
  return r.f;
}
// Reduce across a 16-lane half-row; every lane ends with the result.
__device__ __forceinline__ float half_max16(float v) {
  v = fmaxf(v, dpp_ror<1>(v));
  v = fmaxf(v, dpp_ror<2>(v));
  v = fmaxf(v, dpp_ror<4>(v));
  v = fmaxf(v, dpp_ror<8>(v));
  return v;
}
__device__ __forceinline__ float half_sum16(float v) {
  v += dpp_ror<1>(v);
  v += dpp_ror<2>(v);
  v += dpp_ror<4>(v);
  v += dpp_ror<8>(v);
  return v;
}
// Full wave32 sum: 4 DPP steps within rows + one cross-row shuffle.
__device__ __forceinline__ float wave_sum32(float v) {
  v = half_sum16(v);
  v += __shfl_xor(v, 16, 32);
  return v;
}

// ---------------------------------------------------------------------------
// Kernel: weight convert + transpose f32[K][N] -> f16 Bt[n*K + k]
// ---------------------------------------------------------------------------
__global__ void k_w_to_f16t(const float* __restrict__ W, f16* __restrict__ Wt,
                            int K, int N) {
  int idx = blockIdx.x * blockDim.x + threadIdx.x;  // over N*K, write-contig
  if (idx >= N * K) return;
  int n = idx / K, k = idx - n * K;
  Wt[idx] = (f16)W[(size_t)k * N + n];
}

// ---------------------------------------------------------------------------
// Kernel: mod = silu(t_embed) @ ada_w + ada_b   (4 x 512 x 3072, tiny)
// ---------------------------------------------------------------------------
__global__ void k_modulation(const float* __restrict__ t_embed,
                             const float* __restrict__ ada_w,
                             const float* __restrict__ ada_b,
                             float* __restrict__ mod) {
  int col = blockIdx.x * blockDim.x + threadIdx.x;
  int b   = blockIdx.y;
  if (col >= cfg::MODC) return;
  const float* t = t_embed + b * cfg::DIM;
  float acc = ada_b[col];
#pragma unroll 4
  for (int k = 0; k < cfg::DIM; ++k) {
    float x = t[k];
    acc += (x / (1.f + __expf(-x))) * ada_w[(size_t)k * cfg::MODC + col];
  }
  mod[b * cfg::MODC + col] = acc;
}

// ---------------------------------------------------------------------------
// Kernel: layernorm + adaLN modulate -> f16  (one block per (b,s) row)
// ---------------------------------------------------------------------------
__global__ void k_ln_modulate(const float* __restrict__ inp,
                              const float* __restrict__ gamma,
                              const float* __restrict__ beta,
                              const float* __restrict__ mod,
                              f16* __restrict__ out) {
  const int row = blockIdx.x;          // 0..8191
  const int b   = row >> 11;           // /2048
  const int t   = threadIdx.x;         // 0..255
  const float* x = inp + (size_t)row * cfg::DIM;
  float v0 = x[t], v1 = x[t + 256];
  float s = v0 + v1, sq = v0 * v0 + v1 * v1;
  s  = wave_sum32(s);
  sq = wave_sum32(sq);
  __shared__ float ls[8], lq[8];
  int w = t >> 5, ln = t & 31;
  if (ln == 0) { ls[w] = s; lq[w] = sq; }
  __syncthreads();
  if (t == 0) {
    float a = 0.f, q = 0.f;
#pragma unroll
    for (int i = 0; i < 8; ++i) { a += ls[i]; q += lq[i]; }
    ls[0] = a; lq[0] = q;
  }
  __syncthreads();
  float mean = ls[0] * (1.f / cfg::DIM);
  float var  = lq[0] * (1.f / cfg::DIM) - mean * mean;
  float inv  = rsqrtf(var + 1e-5f);
  const float* mb = mod + b * cfg::MODC;   // [sh_msa | sc_msa | ...]
  int c0 = t, c1 = t + 256;
  float h0 = ((v0 - mean) * inv * gamma[c0] + beta[c0]) * (1.f + mb[512 + c0]) + mb[c0];
  float h1 = ((v1 - mean) * inv * gamma[c1] + beta[c1]) * (1.f + mb[512 + c1]) + mb[c1];
  f16* o = out + (size_t)row * cfg::DIM;
  o[c0] = (f16)h0;
  o[c1] = (f16)h1;
}

// ---------------------------------------------------------------------------
// Kernel: WMMA GEMM  C[MxN] = A[MxK](f16,row) @ Bt[NxK](f16) + bias.
// One wave per 16x64 output strip: the A fragment is reused across 4 WMMAs
// per K-step (4x less A traffic, 4 accumulators). N must be a multiple of 64.
// ---------------------------------------------------------------------------
template <bool OUT_F16>
__global__ void k_wmma_gemm(const f16* __restrict__ A, const f16* __restrict__ Bt,
                            const float* __restrict__ bias, void* __restrict__ C,
                            int M, int N, int K) {
  const int wave = (blockIdx.x * blockDim.x + threadIdx.x) >> 5;
  const int lane = threadIdx.x & 31;
  const int ntn  = N >> 6;                 // 64-wide strips
  const int tiles = (M >> 4) * ntn;
  if (wave >= tiles) return;
  const int tm = (wave / ntn) << 4;
  const int tn = (wave % ntn) << 6;
  const int m = lane & 15, hi = lane >> 4;
  const f16* arow = A  + (size_t)(tm + m) * K;
  const f16* brow = Bt + (size_t)(tn + m) * K;  // +dt*16*K per column tile
  v8f acc[4] = {};
  for (int k0 = 0; k0 < K; k0 += 32) {
    __builtin_prefetch(arow + k0 + 256, 0, 1);  // global_prefetch_b8
    v16h a = load_frag(arow + k0, hi);
#pragma unroll
    for (int dt = 0; dt < 4; ++dt) {
      v16h b = load_frag(brow + (size_t)(dt * 16) * K + k0, hi);
      acc[dt] = wmma_f16(a, b, acc[dt]);
    }
  }
#pragma unroll
  for (int dt = 0; dt < 4; ++dt) {
    const float bn = bias ? bias[tn + dt * 16 + m] : 0.f;
#pragma unroll
    for (int r = 0; r < 8; ++r) {
      float v = acc[dt][r] + bn;
      size_t idx = (size_t)(tm + r + 8 * hi) * N + tn + dt * 16 + m;
      if (OUT_F16) ((f16*)C)[idx] = (f16)v;
      else         ((float*)C)[idx] = v;
    }
  }
}

// ---------------------------------------------------------------------------
// Kernel: per-head RMS norm + stream fusion.
// q = rms(q_c)*w_cq + rms(q_x)*w_xq ; k likewise ; v = v_c + v_x (transposed).
// One wave per head, one block per (b,s) row.
// ---------------------------------------------------------------------------
__global__ void k_qkv_combine(const f16* __restrict__ qkv_c,
                              const f16* __restrict__ qkv_x,
                              const float* __restrict__ rcq, const float* __restrict__ rck,
                              const float* __restrict__ rxq, const float* __restrict__ rxk,
                              f16* __restrict__ Q, f16* __restrict__ Kt,
                              f16* __restrict__ Vt) {
  const int row = blockIdx.x;            // b*SEQ + s
  const int b = row >> 11, s = row & 2047;
  const int h = threadIdx.x >> 5;        // head (8 waves)
  const int lane = threadIdx.x & 31;
  const int d0 = 2 * lane, d1 = d0 + 1;
  const f16* pc = qkv_c + (size_t)row * 1536 + h * 64;
  const f16* px = qkv_x + (size_t)row * 1536 + h * 64;
  float qc0 = (float)pc[d0],        qc1 = (float)pc[d1];
  float kc0 = (float)pc[512 + d0],  kc1 = (float)pc[512 + d1];
  float vc0 = (float)pc[1024 + d0], vc1 = (float)pc[1024 + d1];
  float qx0 = (float)px[d0],        qx1 = (float)px[d1];
  float kx0 = (float)px[512 + d0],  kx1 = (float)px[512 + d1];
  float vx0 = (float)px[1024 + d0], vx1 = (float)px[1024 + d1];
  float iqc = rsqrtf(wave_sum32(qc0 * qc0 + qc1 * qc1) * (1.f / 64.f) + 1e-6f);
  float iqx = rsqrtf(wave_sum32(qx0 * qx0 + qx1 * qx1) * (1.f / 64.f) + 1e-6f);
  float ikc = rsqrtf(wave_sum32(kc0 * kc0 + kc1 * kc1) * (1.f / 64.f) + 1e-6f);
  float ikx = rsqrtf(wave_sum32(kx0 * kx0 + kx1 * kx1) * (1.f / 64.f) + 1e-6f);
  float q0 = qc0 * iqc * rcq[d0] + qx0 * iqx * rxq[d0];
  float q1 = qc1 * iqc * rcq[d1] + qx1 * iqx * rxq[d1];
  float k0 = kc0 * ikc * rck[d0] + kx0 * ikx * rxk[d0];
  float k1 = kc1 * ikc * rck[d1] + kx1 * ikx * rxk[d1];
  const size_t bh = (size_t)(b * cfg::HEADS + h);
  f16* qo = Q  + (bh * cfg::SEQ + s) * 64;
  f16* ko = Kt + (bh * cfg::SEQ + s) * 64;
  qo[d0] = (f16)q0; qo[d1] = (f16)q1;
  ko[d0] = (f16)k0; ko[d1] = (f16)k1;
  f16* vo = Vt + (bh * 64) * cfg::SEQ;   // [bh][d][s]
  vo[(size_t)d0 * cfg::SEQ + s] = (f16)(vc0 + vx0);
  vo[(size_t)d1 * cfg::SEQ + s] = (f16)(vc1 + vx1);
}

// ---------------------------------------------------------------------------
// Kernel: flash attention. One wave owns 16 query rows of one (b,h) and
// streams all 128 key tiles: S = Q·Kᵀ (2 WMMA), online softmax (DPP row
// reductions), O += P·V (4 WMMA, P transposed via zero-padded LDS tile).
// Block = 4 waves.
// ---------------------------------------------------------------------------
__global__ void k_attention(const f16* __restrict__ Q, const f16* __restrict__ Kt,
                            const f16* __restrict__ Vt, f16* __restrict__ O) {
  const int gw   = (blockIdx.x * blockDim.x + threadIdx.x) >> 5;
  const int lane = threadIdx.x & 31;
  const int wib  = threadIdx.x >> 5;
  const int qtiles = cfg::SEQ / 16;          // 128
  const int bh = gw / qtiles;
  const int qt = gw % qtiles;
  const int b = bh >> 3, h = bh & 7;
  const int m = lane & 15, hi = lane >> 4;

  __shared__ f16 plds[4][16 * 32];
  f16* pl = plds[wib];
#pragma unroll
  for (int i = lane; i < 16 * 32; i += 32) pl[i] = (f16)0;  // K-pad stays 0

  const f16* qrow = Q + ((size_t)bh * cfg::SEQ + qt * 16 + m) * 64;
  const v16h qa0 = load_frag(qrow, hi);        // d = 0..31
  const v16h qa1 = load_frag(qrow + 32, hi);   // d = 32..63

  v8f o[4] = {};
  float rm[8], rl[8];
#pragma unroll
  for (int r = 0; r < 8; ++r) { rm[r] = -3.0e38f; rl[r] = 0.f; }
  const float scale = 0.125f;                  // DHEAD^-0.5

  for (int kt = 0; kt < qtiles; ++kt) {
    const f16* krow = Kt + ((size_t)bh * cfg::SEQ + kt * 16 + m) * 64;
    v16h kb0 = load_frag(krow, hi);
    v16h kb1 = load_frag(krow + 32, hi);
    v8f sc = {};
    sc = wmma_f16(qa0, kb0, sc);
    sc = wmma_f16(qa1, kb1, sc);
    // online softmax across key tiles; row (r + 8*hi) lives on one 16-lane half
#pragma unroll
    for (int r = 0; r < 8; ++r) {
      float sv = sc[r] * scale;
      float mx = half_max16(sv);
      float mnew = fmaxf(rm[r], mx);
      float alpha = __expf(rm[r] - mnew);
      rm[r] = mnew;
      float p = __expf(sv - mnew);
      rl[r] = rl[r] * alpha + half_sum16(p);
      sc[r] = p;
#pragma unroll
      for (int dt = 0; dt < 4; ++dt) o[dt][r] *= alpha;
      pl[(r + 8 * hi) * 32 + m] = (f16)p;      // transpose P through LDS
    }
    v16h pa = load_frag(pl + m * 32, hi);      // A = P (16x16, K-padded)
#pragma unroll
    for (int dt = 0; dt < 4; ++dt) {
      const f16* vrow = Vt + ((size_t)bh * 64 + dt * 16 + m) * cfg::SEQ + kt * 16;
      v16h vb = load_frag_k16(vrow, hi);       // only k=0..15 valid
      o[dt] = wmma_f16(pa, vb, o[dt]);
    }
  }
  // write O / l  ->  attnout[(b*SEQ + s)*512 + h*64 + d]
#pragma unroll
  for (int dt = 0; dt < 4; ++dt) {
#pragma unroll
    for (int r = 0; r < 8; ++r) {
      float val = o[dt][r] / rl[r];
      size_t srow = (size_t)(b * cfg::SEQ + qt * 16 + r + 8 * hi);
      O[srow * cfg::DIM + h * 64 + dt * 16 + m] = (f16)val;
    }
  }
}

// ---------------------------------------------------------------------------
// Out-branch epilogues.
// ---------------------------------------------------------------------------
__global__ void k_epilogue1(const float* __restrict__ resid,
                            const float* __restrict__ y1,
                            const float* __restrict__ mod,
                            f16* __restrict__ h2) {
  int i = blockIdx.x * blockDim.x + threadIdx.x;      // over 8192*512
  int col = i & 511;
  int b = i >> 20;                                    // /(2048*512)
  const float* mb = mod + b * cfg::MODC;
  float t = resid[i] + y1[i] * mb[2 * 512 + col];     // g_msa
  float s = t / (1.f + __expf(-t));                   // silu
  h2[i] = (f16)(s * (1.f + mb[4 * 512 + col]) + mb[3 * 512 + col]);
}

__global__ void k_epilogue2(const float* __restrict__ resid,
                            const float* __restrict__ y2,
                            const float* __restrict__ mod,
                            float* __restrict__ out) {
  int i = blockIdx.x * blockDim.x + threadIdx.x;
  int col = i & 511;
  int b = i >> 20;
  out[i] = resid[i] + y2[i] * mod[b * cfg::MODC + 5 * 512 + col];  // g_mlp
}

// ---------------------------------------------------------------------------
// Host launcher.
// ---------------------------------------------------------------------------
extern "C" void kernel_launch(void* const* d_in, const int* in_sizes, int n_in,
                              void* d_out, int out_size, void* d_ws, size_t ws_size,
                              hipStream_t stream) {
  using namespace cfg;
  (void)in_sizes; (void)n_in; (void)out_size; (void)ws_size;

  const float* x      = (const float*)d_in[0];
  const float* c      = (const float*)d_in[1];
  const float* t_emb  = (const float*)d_in[2];
  const float* ln_c_g = (const float*)d_in[3];
  const float* ln_c_b = (const float*)d_in[4];
  const float* ln_x_g = (const float*)d_in[5];
  const float* ln_x_b = (const float*)d_in[6];
  const float* ada_w  = (const float*)d_in[7];
  const float* ada_b  = (const float*)d_in[8];
  const float* lin_c_w = (const float*)d_in[9];
  const float* lin_c_b = (const float*)d_in[10];
  const float* lin_x_w = (const float*)d_in[11];
  const float* lin_x_b = (const float*)d_in[12];
  const float* qkv_c_w = (const float*)d_in[13];
  const float* qkv_x_w = (const float*)d_in[14];
  const float* rms_cq = (const float*)d_in[15];
  const float* rms_ck = (const float*)d_in[16];
  const float* rms_xq = (const float*)d_in[17];
  const float* rms_xk = (const float*)d_in[18];
  const float* outc0_w = (const float*)d_in[19];
  const float* outc0_b = (const float*)d_in[20];
  const float* outc2_w = (const float*)d_in[21];
  const float* outc2_b = (const float*)d_in[22];
  const float* outx0_w = (const float*)d_in[23];
  const float* outx0_b = (const float*)d_in[24];
  const float* outx2_w = (const float*)d_in[25];
  const float* outx2_b = (const float*)d_in[26];

  float* out_x = (float*)d_out;                        // first tuple element
  float* out_c = (float*)d_out + (size_t)ROWS * DIM;   // second

  // workspace bump allocator (256B aligned)
  char* ws = (char*)d_ws;
  size_t off = 0;
  auto alloc = [&](size_t bytes) -> void* {
    off = (off + 255) & ~(size_t)255;
    void* p = ws + off;
    off += bytes;
    return p;
  };
  f16* wt_lin_c = (f16*)alloc((size_t)DIM * DIM * 2);
  f16* wt_lin_x = (f16*)alloc((size_t)DIM * DIM * 2);
  f16* wt_qkv_c = (f16*)alloc((size_t)DIM * 3 * DIM * 2);
  f16* wt_qkv_x = (f16*)alloc((size_t)DIM * 3 * DIM * 2);
  f16* wt_oc0   = (f16*)alloc((size_t)DIM * DIM * 2);
  f16* wt_oc2   = (f16*)alloc((size_t)DIM * DIM * 2);
  f16* wt_ox0   = (f16*)alloc((size_t)DIM * DIM * 2);
  f16* wt_ox2   = (f16*)alloc((size_t)DIM * DIM * 2);
  float* mod    = (float*)alloc((size_t)BATCH * MODC * 4);
  f16* h_c   = (f16*)alloc((size_t)ROWS * DIM * 2);
  f16* h_x   = (f16*)alloc((size_t)ROWS * DIM * 2);
  f16* h2_c  = (f16*)alloc((size_t)ROWS * DIM * 2);
  f16* h2_x  = (f16*)alloc((size_t)ROWS * DIM * 2);
  f16* qkv_c = (f16*)alloc((size_t)ROWS * 3 * DIM * 2);
  f16* qkv_x = (f16*)alloc((size_t)ROWS * 3 * DIM * 2);
  f16* qbuf  = (f16*)alloc((size_t)ROWS * DIM * 2);
  f16* kbuf  = (f16*)alloc((size_t)ROWS * DIM * 2);
  f16* vtbuf = (f16*)alloc((size_t)ROWS * DIM * 2 + 256);
  f16* attno = (f16*)alloc((size_t)ROWS * DIM * 2);
  float* y1  = (float*)alloc((size_t)ROWS * DIM * 4);
  f16* h2mlp = (f16*)alloc((size_t)ROWS * DIM * 2);

  auto wconv = [&](const float* W, f16* Wt, int K, int N) {
    int n = K * N;
    k_w_to_f16t<<<(n + 255) / 256, 256, 0, stream>>>(W, Wt, K, N);
  };
  wconv(lin_c_w, wt_lin_c, DIM, DIM);
  wconv(lin_x_w, wt_lin_x, DIM, DIM);
  wconv(qkv_c_w, wt_qkv_c, DIM, 3 * DIM);
  wconv(qkv_x_w, wt_qkv_x, DIM, 3 * DIM);
  wconv(outc0_w, wt_oc0, DIM, DIM);
  wconv(outc2_w, wt_oc2, DIM, DIM);
  wconv(outx0_w, wt_ox0, DIM, DIM);
  wconv(outx2_w, wt_ox2, DIM, DIM);

  k_modulation<<<dim3((MODC + 255) / 256, BATCH), 256, 0, stream>>>(t_emb, ada_w, ada_b, mod);

  k_ln_modulate<<<ROWS, 256, 0, stream>>>(c, ln_c_g, ln_c_b, mod, h_c);
  k_ln_modulate<<<ROWS, 256, 0, stream>>>(x, ln_x_g, ln_x_b, mod, h_x);

  auto gemm_f16 = [&](const f16* A, const f16* Bt, const float* bias, f16* C,
                      int M, int N, int K) {
    int tiles = (M / 16) * (N / 64);   // 16x64 strip per wave
    k_wmma_gemm<true><<<(tiles + 7) / 8, 256, 0, stream>>>(A, Bt, bias, C, M, N, K);
  };
  auto gemm_f32 = [&](const f16* A, const f16* Bt, const float* bias, float* C,
                      int M, int N, int K) {
    int tiles = (M / 16) * (N / 64);
    k_wmma_gemm<false><<<(tiles + 7) / 8, 256, 0, stream>>>(A, Bt, bias, C, M, N, K);
  };

  gemm_f16(h_c, wt_lin_c, lin_c_b, h2_c, ROWS, DIM, DIM);
  gemm_f16(h_x, wt_lin_x, lin_x_b, h2_x, ROWS, DIM, DIM);
  gemm_f16(h2_c, wt_qkv_c, nullptr, qkv_c, ROWS, 3 * DIM, DIM);
  gemm_f16(h2_x, wt_qkv_x, nullptr, qkv_x, ROWS, 3 * DIM, DIM);

  k_qkv_combine<<<ROWS, 256, 0, stream>>>(qkv_c, qkv_x, rms_cq, rms_ck, rms_xq,
                                          rms_xk, qbuf, kbuf, vtbuf);

  // 32 (b,h) * 128 qtiles = 4096 waves; 4 waves/block -> 1024 blocks
  k_attention<<<(BATCH * HEADS * (SEQ / 16)) / 4, 128, 0, stream>>>(qbuf, kbuf,
                                                                    vtbuf, attno);

  const int nelem = ROWS * DIM;
  // stream c
  gemm_f32(attno, wt_oc0, outc0_b, y1, ROWS, DIM, DIM);
  k_epilogue1<<<nelem / 256, 256, 0, stream>>>(c, y1, mod, h2mlp);
  gemm_f32(h2mlp, wt_oc2, outc2_b, y1, ROWS, DIM, DIM);
  k_epilogue2<<<nelem / 256, 256, 0, stream>>>(c, y1, mod, out_c);
  // stream x
  gemm_f32(attno, wt_ox0, outx0_b, y1, ROWS, DIM, DIM);
  k_epilogue1<<<nelem / 256, 256, 0, stream>>>(x, y1, mod, h2mlp);
  gemm_f32(h2mlp, wt_ox2, outx2_b, y1, ROWS, DIM, DIM);
  k_epilogue2<<<nelem / 256, 256, 0, stream>>>(x, y1, mod, out_x);
}
